// TimeSelfattention_34411277976416
// MI455X (gfx1250) — matrix operand order repaired and checked by reading, not compiled, not run on hardware
//
#include <hip/hip_runtime.h>
#include <hip/hip_bf16.h>
#include <math.h>

// Linear attention: B=8, C=512, N=4096, H=8, dh=64.
//   Ksm = softmax over tokens (rows of K[b,c,:])            -> kernel 1 stats
//   kvs[bh][d][e] = sum_n Ksm[d,n] * V[e,n]                 -> kernel 2 (WMMA f16, split-N + atomics)
//   out[b,h*64+e,n] = sum_d Qsm[d,n] * kvs[d,e]             -> kernel 3 (WMMA f16)
// Memory-bound: ~350 MB traffic @ 23.3 TB/s ~ 15 us; ~262K v_wmma_f32_16x16x32_f16 total.

typedef __attribute__((ext_vector_type(16))) _Float16 v16h;
typedef __attribute__((ext_vector_type(8)))  float    v8f;

#define NB    8
#define CDIM  512
#define NTOK  4096
#define NH    8
#define DH    64
#define BH    64        // NB*NH
// workspace layout (floats): [0,8192) = (rowmax,rowsum) per (b,c); [8192, 8192+262144) = kvs
#define WS_STATS 0
#define WS_KVS   8192

// ---------------- Kernel 1: K row softmax stats + zero kvs accumulator ----------------
__global__ void k_rowstats(const float* __restrict__ k, float* __restrict__ ws) {
  __shared__ float red[256];
  const int row = blockIdx.x;                       // row = b*512 + c, 4096 rows
  const int t = threadIdx.x;
  const float* kr = k + (size_t)row * NTOK;
  float vals[16];
  float m = -INFINITY;
  #pragma unroll
  for (int i = 0; i < 16; ++i) { vals[i] = kr[t + i * 256]; m = fmaxf(m, vals[i]); }
  red[t] = m; __syncthreads();
  for (int s = 128; s > 0; s >>= 1) { if (t < s) red[t] = fmaxf(red[t], red[t + s]); __syncthreads(); }
  m = red[0]; __syncthreads();
  float sum = 0.f;
  #pragma unroll
  for (int i = 0; i < 16; ++i) sum += __expf(vals[i] - m);
  red[t] = sum; __syncthreads();
  for (int s = 128; s > 0; s >>= 1) { if (t < s) red[t] += red[t + s]; __syncthreads(); }
  if (t == 0) { ws[WS_STATS + row * 2] = m; ws[WS_STATS + row * 2 + 1] = red[0]; }
  if (t < 64) ws[WS_KVS + (size_t)row * 64 + t] = 0.f;   // 4096 blocks * 64 = full kvs
}

// ---------------- Kernel 2: kvs = Ksm @ V^T  (split over token chunks, atomic accumulate) ----------------
#define K2_TOKS 512      // tokens per block
#define K2_CH   32       // tokens per WMMA k-step
__global__ void k_kvstate(const float* __restrict__ kg, const float* __restrict__ vg,
                          float* __restrict__ ws) {
  __shared__ _Float16 ksm[64][K2_CH + 4];
  __shared__ _Float16 vsm[64][K2_CH + 4];
  __shared__ float st[64][2];
  const int bh = blockIdx.y;
  const int b = bh >> 3, h = bh & 7;
  const int n_start = blockIdx.x * K2_TOKS;
  const int t = threadIdx.x, lane = t & 31, w = t >> 5;
  const size_t base = ((size_t)(b * CDIM + h * DH)) * NTOK;
  if (t < 64) {
    const int row = b * CDIM + h * DH + t;
    st[t][0] = ws[WS_STATS + row * 2];
    st[t][1] = 1.0f / ws[WS_STATS + row * 2 + 1];
  }
  __syncthreads();
  const int dt  = w >> 1;          // output d-tile 0..3
  const int et0 = (w & 1) * 2;     // output e-tiles {et0, et0+1}
  const int half = lane >> 4, mrow = lane & 15;
  v8f c0 = {}; v8f c1 = {};

  for (int ch = 0; ch < K2_TOKS / K2_CH; ++ch) {
    const int n0 = n_start + ch * K2_CH;
    __syncthreads();
    // load 64x32 tiles of K (softmaxed) and V; wave w loads rows w, w+8, ...
    #pragma unroll
    for (int i = 0; i < 8; ++i) {
      const int d = i * 8 + w;
      const float kv = kg[base + (size_t)d * NTOK + n0 + lane];
      ksm[d][lane] = (_Float16)(__expf(kv - st[d][0]) * st[d][1]);
      vsm[d][lane] = (_Float16)vg[base + (size_t)d * NTOK + n0 + lane];
    }
    __syncthreads();
    // A fragment (Ksm tile, 16x32, M=d K=n); ISA layout: lanes<16 K=0..7/16..23, lanes>=16 K=8..15/24..31
    v16h a;
    {
      const int d = dt * 16 + mrow;
      #pragma unroll
      for (int vv = 0; vv < 4; ++vv) {
        const int kk  = half * 8 + 2 * vv;
        const int kk2 = 16 + half * 8 + 2 * vv;
        a[2 * vv]           = ksm[d][kk];
        a[2 * vv + 1]       = ksm[d][kk + 1];
        a[2 * (vv + 4)]     = ksm[d][kk2];
        a[2 * (vv + 4) + 1] = ksm[d][kk2 + 1];
      }
    }
    // B fragments (V^T, 32x16, K=n N=e); lanes<16 K=0..7/16..23, lanes>=16 K=8..15/24..31
    #pragma unroll
    for (int tt = 0; tt < 2; ++tt) {
      const int e = (et0 + tt) * 16 + mrow;
      v16h bf;
      #pragma unroll
      for (int j = 0; j < 8; ++j) {
        const int kk = (j >> 2) * 8 + half * 8 + 2 * j;
        bf[2 * j]     = vsm[e][kk];
        bf[2 * j + 1] = vsm[e][kk + 1];
      }
      if (tt == 0)
        c0 = __builtin_amdgcn_wmma_f32_16x16x32_f16(false, a, false, bf, (short)0, c0, false, false);
      else
        c1 = __builtin_amdgcn_wmma_f32_16x16x32_f16(false, a, false, bf, (short)0, c1, false, false);
    }
  }
  // accumulate partial kvs tile into workspace (D layout: vgpr r -> M=r(+8 for upper lanes), N=lane%16)
  float* kvbase = ws + WS_KVS + (size_t)bh * 4096;
  #pragma unroll
  for (int r = 0; r < 8; ++r) {
    const int d = dt * 16 + r + half * 8;
    atomicAdd(&kvbase[d * 64 + et0 * 16 + mrow],        c0[r]);
    atomicAdd(&kvbase[d * 64 + (et0 + 1) * 16 + mrow],  c1[r]);
  }
}

// ---------------- Kernel 3: out = kvs^T @ Qsm ----------------
#define TN 256           // tokens per block
__global__ void k_output(const float* __restrict__ qg, const float* __restrict__ ws,
                         float* __restrict__ out) {
  __shared__ _Float16 kvT[64][64];       // [e][d] = kvs[d][e], f16
  __shared__ _Float16 qsm[TN][66];       // [n_local][d], f16, padded rows
  const int bh = blockIdx.y;
  const int b = bh >> 3, h = bh & 7;
  const int n0 = blockIdx.x * TN;
  const int t = threadIdx.x, lane = t & 31, w = t >> 5;

  // stage kvs transposed as f16 into LDS
  const float* kv = ws + WS_KVS + (size_t)bh * 4096;
  #pragma unroll
  for (int i = 0; i < 16; ++i) {
    const int idx = t + i * 256;
    kvT[idx & 63][idx >> 6] = (_Float16)kv[idx];
  }
  // Q column softmax: thread t owns token n0+t. Pass 1 max, pass 2 exp+sum (L2-hot reread), rescale.
  const size_t qbase = ((size_t)(b * CDIM + h * DH)) * NTOK + n0 + t;
  float m = -INFINITY;
  for (int d = 0; d < 64; ++d) m = fmaxf(m, qg[qbase + (size_t)d * NTOK]);
  float s = 0.f;
  for (int d = 0; d < 64; ++d) {
    const float e = __expf(qg[qbase + (size_t)d * NTOK] - m);
    s += e;
    qsm[t][d] = (_Float16)e;
  }
  const float inv = 1.0f / s;
  for (int d = 0; d < 64; ++d) qsm[t][d] = (_Float16)((float)qsm[t][d] * inv);
  __syncthreads();

  // GEMM: A = kvs^T (M=e, K=d), B = Qsm (K=d, N=n). Wave w: e-tile w%4, n-tiles (w/4)*8 .. +7.
  const int et = w & 3;
  const int ntbase = (w >> 2) * 8;
  const int half = lane >> 4, mrow = lane & 15;
  v16h a0, a1;
  #pragma unroll
  for (int kf = 0; kf < 2; ++kf) {
    v16h a;
    const int e = et * 16 + mrow;
    #pragma unroll
    for (int vv = 0; vv < 4; ++vv) {
      const int kk  = kf * 32 + half * 8 + 2 * vv;
      const int kk2 = kf * 32 + 16 + half * 8 + 2 * vv;
      a[2 * vv]           = kvT[e][kk];
      a[2 * vv + 1]       = kvT[e][kk + 1];
      a[2 * (vv + 4)]     = kvT[e][kk2];
      a[2 * (vv + 4) + 1] = kvT[e][kk2 + 1];
    }
    if (kf == 0) a0 = a; else a1 = a;
  }
  #pragma unroll
  for (int j = 0; j < 8; ++j) {
    const int nt = ntbase + j;
    const int nl = nt * 16 + mrow;
    v8f c = {};
    #pragma unroll
    for (int kf = 0; kf < 2; ++kf) {
      v16h bf;
      #pragma unroll
      for (int jj = 0; jj < 8; ++jj) {
        const int kk = kf * 32 + (jj >> 2) * 8 + half * 8 + 2 * jj;
        bf[2 * jj]     = qsm[nl][kk];
        bf[2 * jj + 1] = qsm[nl][kk + 1];
      }
      v16h a = (kf == 0) ? a0 : a1;
      c = __builtin_amdgcn_wmma_f32_16x16x32_f16(false, a, false, bf, (short)0, c, false, false);
    }
    const int n = n0 + nt * 16 + mrow;
    #pragma unroll
    for (int r = 0; r < 8; ++r) {
      const int e = et * 16 + r + half * 8;
      out[((size_t)(b * CDIM + h * DH + e)) * NTOK + n] = c[r];
    }
  }
}

extern "C" void kernel_launch(void* const* d_in, const int* in_sizes, int n_in,
                              void* d_out, int out_size, void* d_ws, size_t ws_size,
                              hipStream_t stream) {
  const float* q = (const float*)d_in[0];
  const float* k = (const float*)d_in[1];
  const float* v = (const float*)d_in[2];
  float* out = (float*)d_out;
  float* ws  = (float*)d_ws;          // needs (8192 + 64*4096)*4 ~= 1.06 MB

  k_rowstats<<<4096, 256, 0, stream>>>(k, ws);
  k_kvstate<<<dim3(NTOK / K2_TOKS, BH), 256, 0, stream>>>(k, v, ws);
  k_output<<<dim3(NTOK / TN, BH), 256, 0, stream>>>(q, ws, out);
}
